// Model_75058848465345
// MI455X (gfx1250) — compile-verified, matrix operand order checked by reading, not run on hardware
//
#include <hip/hip_runtime.h>
#include <math.h>

// ---------------------------------------------------------------------------
// Types for CDNA5 WMMA (wave32): v_wmma_f32_16x16x32_f16
// ---------------------------------------------------------------------------
typedef _Float16 h8  __attribute__((ext_vector_type(8)));
typedef _Float16 h16 __attribute__((ext_vector_type(16)));
typedef float    f8  __attribute__((ext_vector_type(8)));

#define B_    128
#define L_    512
#define C_    321
#define PRED_ 336
#define N_    128
#define H_    512
#define BC_   (B_ * C_)        // 41088  (multiple of 16)
#define R_    (BC_ * 4)        // 164352 (multiple of 16)

// 16-bit A/B fragment layout (CDNA5 ISA 7.12.2):
//   lane l: row/col = l%16 ; halfs 0..7  = K[kbase .. kbase+7]
//                            halfs 8..15 = K[kbase+16 .. kbase+23]
//   kbase = kk*32 + (l>=16 ? 8 : 0)
// => two contiguous 16-byte runs per lane; source buffers are stored
//    K-contiguous per row (A) / per output-column (B, i.e. W transposed).
__device__ __forceinline__ h16 load_frag(const _Float16* row, int kbase) {
  h8 lo = *(const h8*)(row + kbase);
  h8 hi = *(const h8*)(row + kbase + 16);
  h16 f;
#pragma unroll
  for (int i = 0; i < 8; ++i) { f[i] = lo[i]; f[i + 8] = hi[i]; }
  return f;
}

__device__ __forceinline__ f8 wmma16(h16 a, h16 b, f8 c) {
  return __builtin_amdgcn_wmma_f32_16x16x32_f16(false, a, false, b,
                                                (short)0, c, false, false);
}

// Branch-free tanh-form GELU:
//   u = sqrt(2/pi) * (x + 0.044715 x^3)
//   gelu = 0.5 x (1 + tanh u) = x * sigmoid(2u) = x * (1 - rcp(exp2(2u*log2e)+1))
// Lowers to straight-line VALU with one v_exp_f32 and one v_rcp_f32
// (no EXEC-divergent libm erf expansion inside the WMMA epilogue).
__device__ __forceinline__ float gelu_fast(float x) {
  float u = x * (0.7978845608028654f + 0.0356774081363001f * x * x);
  float e = __builtin_amdgcn_exp2f(u * 2.8853900817779268f);  // exp(2u)
  return x * (1.0f - __builtin_amdgcn_rcpf(e + 1.0f));
}

// ---------------------------------------------------------------------------
// Weight prep: fp32 [rows, cols] -> fp16 transposed [cols, rows] (K-contig)
// ---------------------------------------------------------------------------
__global__ __launch_bounds__(256) void transpose_to_h(const float* __restrict__ in,
                                                      _Float16* __restrict__ out,
                                                      int rows, int cols) {
  int idx = blockIdx.x * blockDim.x + threadIdx.x;
  if (idx >= rows * cols) return;
  int r = idx / cols, c = idx - r * cols;
  out[(size_t)c * rows + r] = (_Float16)in[idx];
}

// ---------------------------------------------------------------------------
// RevIN stats + normalize + K=25 moving-average decomposition.
// Writes: mean/std [BC], trend fp16 [bc,512], seasonal fp16 in split layout
//         xs[(bc*4+s)*128 + n]  (l = n*4+s).
// Block: 256 thr = 16 channels x 16 time-slices of 32; grid (21, B).
// ---------------------------------------------------------------------------
__global__ __launch_bounds__(256) void stats_decomp_kernel(
    const float* __restrict__ x, const float* __restrict__ rw,
    const float* __restrict__ rb, float* __restrict__ meanb,
    float* __restrict__ stdb, _Float16* __restrict__ trendb,
    _Float16* __restrict__ xsb) {
  __shared__ float sx[16][521];
  __shared__ float red[2][16][16];
  __shared__ float stats[2][16];
  const int b  = blockIdx.y;
  const int tc = threadIdx.x & 15;
  const int tl = threadIdx.x >> 4;  // 0..15
  const int c  = blockIdx.x * 16 + tc;
  const bool valid = (c < C_);
  const int lo = tl * 32;

  float s = 0.f, s2 = 0.f;
  for (int i = 0; i < 32; ++i) {
    int l = lo + i;
    float v = valid ? x[((size_t)b * L_ + l) * C_ + c] : 0.f;
    sx[tc][l] = v;
    s += v;
    s2 += v * v;
  }
  red[0][tl][tc] = s;
  red[1][tl][tc] = s2;
  __syncthreads();
  if (tl == 0) {
    float ss = 0.f, ss2 = 0.f;
#pragma unroll
    for (int i = 0; i < 16; ++i) { ss += red[0][i][tc]; ss2 += red[1][i][tc]; }
    float mean = ss * (1.0f / L_);
    float var  = ss2 * (1.0f / L_) - mean * mean;
    float stdv = sqrtf(var + 1e-5f);
    stats[0][tc] = mean;
    stats[1][tc] = 1.0f / stdv;
    if (valid) {
      meanb[(size_t)b * C_ + c] = mean;
      stdb[(size_t)b * C_ + c]  = stdv;
    }
  }
  __syncthreads();
  const float mean = stats[0][tc];
  const float rstd = stats[1][tc];
  const float w  = valid ? rw[c] : 1.f;
  const float bb = valid ? rb[c] : 0.f;
  for (int i = 0; i < 32; ++i) {
    int l = lo + i;
    sx[tc][l] = (sx[tc][l] - mean) * rstd * w + bb;
  }
  __syncthreads();

  // moving average width 25 with replicate padding == clamped indices
  float win = 0.f;
#pragma unroll
  for (int d = -12; d <= 12; ++d) {
    int idx = lo + d;
    idx = idx < 0 ? 0 : (idx > L_ - 1 ? L_ - 1 : idx);
    win += sx[tc][idx];
  }
  const size_t bc = (size_t)b * C_ + c;
  for (int i = 0; i < 32; ++i) {
    int l = lo + i;
    float trend = win * (1.0f / 25.0f);
    float sea = sx[tc][l] - trend;
    if (valid) {
      trendb[bc * L_ + l] = (_Float16)trend;
      xsb[(bc * 4 + (l & 3)) * (size_t)N_ + (l >> 2)] = (_Float16)sea;
    }
    int add = l + 13; add = add > L_ - 1 ? L_ - 1 : add;
    int sub = l - 12; sub = sub < 0 ? 0 : sub;
    win += sx[tc][add] - sx[tc][sub];
  }
}

// ---------------------------------------------------------------------------
// Fused 4-layer MLP: 128 ->(gelu) 512 -> 128 ->(gelu) 512 -> 128.
// One wave per 16-row tile; hidden activations stay in per-wave LDS,
// which also performs the D-layout -> A-layout transpose between layers.
// Output overwrites xs in the inverse-interleave (sea[bc,512]) layout.
// ---------------------------------------------------------------------------
__global__ __launch_bounds__(64) void mlp_kernel(
    const _Float16* xs, const _Float16* __restrict__ W11t,
    const _Float16* __restrict__ W12t, const _Float16* __restrict__ W21t,
    const _Float16* __restrict__ W22t, const float* __restrict__ b11,
    const float* __restrict__ b12, const float* __restrict__ b21,
    const float* __restrict__ b22, _Float16* sea) {
  __shared__ __align__(16) _Float16 hidA[2][16][520];  // 16 x 512 (+pad)
  __shared__ __align__(16) _Float16 hidB[2][16][136];  // 16 x 128 (+pad)
  const int warp = threadIdx.x >> 5;
  const int lane = threadIdx.x & 31;
  const int m  = lane & 15;
  const int hs = lane >> 4;        // 0/1: which K-half / which M-half
  const int ks = hs * 8;
  const int row0 = (blockIdx.x * 2 + warp) * 16;

  // ---- layer 1: [16x128] @ W11 [128x512], +b11, GELU ----
  h16 aA[4];
  {
    const _Float16* ar = xs + (size_t)(row0 + m) * N_;
#pragma unroll
    for (int kk = 0; kk < 4; ++kk) aA[kk] = load_frag(ar, kk * 32 + ks);
  }
  for (int j = 0; j < 32; ++j) {
    const _Float16* br = W11t + (size_t)(j * 16 + m) * N_;
    __builtin_prefetch(W11t + (size_t)((j + 1) * 16 + m) * N_, 0, 1);
    f8 acc = {};
#pragma unroll
    for (int kk = 0; kk < 4; ++kk)
      acc = wmma16(aA[kk], load_frag(br, kk * 32 + ks), acc);
    float bias = b11[j * 16 + m];
#pragma unroll
    for (int r = 0; r < 8; ++r)
      hidA[warp][r + 8 * hs][j * 16 + m] = (_Float16)gelu_fast(acc[r] + bias);
  }

  // ---- layer 2: [16x512] @ W12 [512x128], +b12 ----
  for (int j = 0; j < 8; ++j) {
    const _Float16* br = W12t + (size_t)(j * 16 + m) * H_;
    const _Float16* ar = &hidA[warp][m][0];
    f8 acc = {};
#pragma unroll 4
    for (int kk = 0; kk < 16; ++kk)
      acc = wmma16(load_frag(ar, kk * 32 + ks), load_frag(br, kk * 32 + ks), acc);
    float bias = b12[j * 16 + m];
#pragma unroll
    for (int r = 0; r < 8; ++r)
      hidB[warp][r + 8 * hs][j * 16 + m] = (_Float16)(acc[r] + bias);
  }

  // ---- layer 3: [16x128] @ W21 [128x512], +b21, GELU ----
  {
    const _Float16* ar = &hidB[warp][m][0];
#pragma unroll
    for (int kk = 0; kk < 4; ++kk) aA[kk] = load_frag(ar, kk * 32 + ks);
  }
  for (int j = 0; j < 32; ++j) {
    const _Float16* br = W21t + (size_t)(j * 16 + m) * N_;
    __builtin_prefetch(W21t + (size_t)((j + 1) * 16 + m) * N_, 0, 1);
    f8 acc = {};
#pragma unroll
    for (int kk = 0; kk < 4; ++kk)
      acc = wmma16(aA[kk], load_frag(br, kk * 32 + ks), acc);
    float bias = b21[j * 16 + m];
#pragma unroll
    for (int r = 0; r < 8; ++r)
      hidA[warp][r + 8 * hs][j * 16 + m] = (_Float16)gelu_fast(acc[r] + bias);
  }

  // ---- layer 4: [16x512] @ W22 [512x128], +b22, inverse interleave ----
  for (int j = 0; j < 8; ++j) {
    const _Float16* br = W22t + (size_t)(j * 16 + m) * H_;
    const _Float16* ar = &hidA[warp][m][0];
    f8 acc = {};
#pragma unroll 4
    for (int kk = 0; kk < 16; ++kk)
      acc = wmma16(load_frag(ar, kk * 32 + ks), load_frag(br, kk * 32 + ks), acc);
    float bias = b22[j * 16 + m];
#pragma unroll
    for (int r = 0; r < 8; ++r) {
      int g = row0 + r + 8 * hs;      // global MLP row = bc*4 + s
      int bc = g >> 2, s = g & 3;
      int n = j * 16 + m;
      sea[(size_t)bc * L_ + n * 4 + s] = (_Float16)(acc[r] + bias);
    }
  }
}

// ---------------------------------------------------------------------------
// Heads: out[bc, p] = trend[bc,:] @ W_tr[:,p] + sea[bc,:] @ W_se[:,p] + biases
// followed by fused RevIN denorm and scatter to [B, PRED, C].
// One 16x16 tile per wave, K = 512 for each of the two products.
// ---------------------------------------------------------------------------
__global__ __launch_bounds__(128) void head_kernel(
    const _Float16* __restrict__ trendb, const _Float16* __restrict__ seab,
    const _Float16* __restrict__ Wtrt, const _Float16* __restrict__ Wset,
    const float* __restrict__ btr, const float* __restrict__ bse,
    const float* __restrict__ rw, const float* __restrict__ rb,
    const float* __restrict__ meanb, const float* __restrict__ stdb,
    float* __restrict__ out) {
  const int wv = blockIdx.x * 4 + (threadIdx.x >> 5);
  const int lane = threadIdx.x & 31;
  const int m  = lane & 15;
  const int ks = (lane >> 4) * 8;
  const int bc0 = (wv / 21) * 16;
  const int p0  = (wv % 21) * 16;

  f8 acc = {};
  {
    const _Float16* ar = trendb + (size_t)(bc0 + m) * L_;
    const _Float16* br = Wtrt + (size_t)(p0 + m) * L_;
#pragma unroll 4
    for (int kk = 0; kk < 16; ++kk)
      acc = wmma16(load_frag(ar, kk * 32 + ks), load_frag(br, kk * 32 + ks), acc);
  }
  {
    const _Float16* ar = seab + (size_t)(bc0 + m) * L_;
    const _Float16* br = Wset + (size_t)(p0 + m) * L_;
#pragma unroll 4
    for (int kk = 0; kk < 16; ++kk)
      acc = wmma16(load_frag(ar, kk * 32 + ks), load_frag(br, kk * 32 + ks), acc);
  }
  const int p = p0 + m;
  const float bias = btr[p] + bse[p];
#pragma unroll
  for (int r = 0; r < 8; ++r) {
    int M = bc0 + r + 8 * (lane >> 4);
    int b = M / C_, c = M - b * C_;
    float v = acc[r] + bias;
    v = (v - rb[c]) / (rw[c] + 1e-10f);   // EPS*EPS
    v = v * stdb[M] + meanb[M];
    out[((size_t)b * PRED_ + p) * C_ + c] = v;
  }
}

// ---------------------------------------------------------------------------
// Host launcher
// ---------------------------------------------------------------------------
extern "C" void kernel_launch(void* const* d_in, const int* in_sizes, int n_in,
                              void* d_out, int out_size, void* d_ws,
                              size_t ws_size, hipStream_t stream) {
  const float* x   = (const float*)d_in[0];
  const float* rw  = (const float*)d_in[1];
  const float* rb  = (const float*)d_in[2];
  const float* W11 = (const float*)d_in[3];
  const float* b11 = (const float*)d_in[4];
  const float* W12 = (const float*)d_in[5];
  const float* b12 = (const float*)d_in[6];
  const float* W21 = (const float*)d_in[7];
  const float* b21 = (const float*)d_in[8];
  const float* W22 = (const float*)d_in[9];
  const float* b22 = (const float*)d_in[10];
  const float* Wtr = (const float*)d_in[11];
  const float* btr = (const float*)d_in[12];
  const float* Wse = (const float*)d_in[13];
  const float* bse = (const float*)d_in[14];

  char* p = (char*)d_ws;
  auto alloc = [&](size_t bytes) {
    void* r = (void*)p;
    p += (bytes + 255) & ~(size_t)255;
    return r;
  };
  float*    meanb  = (float*)alloc((size_t)BC_ * 4);
  float*    stdb   = (float*)alloc((size_t)BC_ * 4);
  _Float16* W11t   = (_Float16*)alloc((size_t)N_ * H_ * 2);
  _Float16* W12t   = (_Float16*)alloc((size_t)N_ * H_ * 2);
  _Float16* W21t   = (_Float16*)alloc((size_t)N_ * H_ * 2);
  _Float16* W22t   = (_Float16*)alloc((size_t)N_ * H_ * 2);
  _Float16* Wtrt   = (_Float16*)alloc((size_t)L_ * PRED_ * 2);
  _Float16* Wset   = (_Float16*)alloc((size_t)L_ * PRED_ * 2);
  _Float16* trendb = (_Float16*)alloc((size_t)BC_ * L_ * 2);
  _Float16* xsb    = (_Float16*)alloc((size_t)R_ * N_ * 2);  // aliased as sea

  // --- weight prep (fp16, K-contiguous B layout) ---
  int nw = N_ * H_;  // 65536
  transpose_to_h<<<(nw + 255) / 256, 256, 0, stream>>>(W11, W11t, N_, H_);
  transpose_to_h<<<(nw + 255) / 256, 256, 0, stream>>>(W12, W12t, H_, N_);
  transpose_to_h<<<(nw + 255) / 256, 256, 0, stream>>>(W21, W21t, N_, H_);
  transpose_to_h<<<(nw + 255) / 256, 256, 0, stream>>>(W22, W22t, H_, N_);
  int nh = L_ * PRED_;  // 172032
  transpose_to_h<<<(nh + 255) / 256, 256, 0, stream>>>(Wtr, Wtrt, L_, PRED_);
  transpose_to_h<<<(nh + 255) / 256, 256, 0, stream>>>(Wse, Wset, L_, PRED_);

  // --- stats + normalize + decomposition ---
  stats_decomp_kernel<<<dim3(21, B_), 256, 0, stream>>>(x, rw, rb, meanb, stdb,
                                                        trendb, xsb);

  // --- fused seasonal MLP (R_/16 = 10272 tiles, 2 waves/block) ---
  mlp_kernel<<<R_ / 32, 64, 0, stream>>>(xsb, W11t, W12t, W21t, W22t, b11, b12,
                                         b21, b22, xsb);

  // --- heads + denorm ((BC_/16)*(PRED_/16) = 53928 tiles, 4 waves/block) ---
  head_kernel<<<(BC_ / 16) * (PRED_ / 16) / 4, 128, 0, stream>>>(
      trendb, xsb, Wtrt, Wset, btr, bse, rw, rb, meanb, stdb, (float*)d_out);
}
